// Attention_19688130085065
// MI455X (gfx1250) — matrix operand order repaired and checked by reading, not compiled, not run on hardware
//
#include <hip/hip_runtime.h>

// ---------------------------------------------------------------------------
// Problem constants
// ---------------------------------------------------------------------------
#define BATCH   4
#define SEQ     2048
#define DIM     512
#define HEADS   8
#define DHEAD   64
#define QKV3    1536
#define MROWS   (BATCH * SEQ)   // 8192
#define LN_EPS  1e-5f
#define MASK_FILL 1e-8f
#define LPITCH  72              // bf16 LDS row pitch (144B: 16B-aligned, bank-spread)

typedef __bf16 v16bf __attribute__((ext_vector_type(16)));
typedef float  v8f   __attribute__((ext_vector_type(8)));
typedef unsigned short v8us  __attribute__((ext_vector_type(8)));
typedef unsigned short v16us __attribute__((ext_vector_type(16)));

// D(16x16,f32) += A(16x32,bf16) x B(32x16,bf16)
__device__ __forceinline__ v8f wmma_bf16(v16us a, v16us b, v8f c) {
  return __builtin_amdgcn_wmma_f32_16x16x32_bf16(
      false, __builtin_bit_cast(v16bf, a), false, __builtin_bit_cast(v16bf, b),
      (short)0, c, false, false);
}

__device__ __forceinline__ v16us mk16(v8us a, v8us b) {
  return __builtin_shufflevector(a, b, 0, 1, 2, 3, 4, 5, 6, 7,
                                 8, 9, 10, 11, 12, 13, 14, 15);
}

// f32 -> bf16 (RNE) hi/lo split: hi + lo reconstructs f to ~2^-17 rel.
__device__ __forceinline__ unsigned short bf_hi(float f) {
  unsigned int u = __float_as_uint(f);
  return (unsigned short)((u + 0x7FFFu + ((u >> 16) & 1u)) >> 16);
}
__device__ __forceinline__ float bf_f32(unsigned short h) {
  return __uint_as_float((unsigned int)h << 16);
}
__device__ __forceinline__ void bf_split(float f, unsigned short& hi, unsigned short& lo) {
  hi = bf_hi(f);
  lo = bf_hi(f - bf_f32(hi));
}

// ---------------------------------------------------------------------------
// Kernel 0: LayerNorm statistics (one wave per row)
// ---------------------------------------------------------------------------
__global__ __launch_bounds__(256) void ln_stats_kernel(
    const float* __restrict__ x, float* __restrict__ mu, float* __restrict__ rstd) {
  int row  = blockIdx.x * 8 + (threadIdx.x >> 5);
  int lane = threadIdx.x & 31;
  const float* xr = x + (size_t)row * DIM;
  float s = 0.f, s2 = 0.f;
#pragma unroll
  for (int i = 0; i < 4; ++i) {
    float4 v = *(const float4*)(xr + lane * 4 + i * 128);
    s  += v.x + v.y + v.z + v.w;
    s2 += v.x * v.x + v.y * v.y + v.z * v.z + v.w * v.w;
  }
#pragma unroll
  for (int m = 16; m >= 1; m >>= 1) {
    s  += __shfl_xor(s,  m, 32);
    s2 += __shfl_xor(s2, m, 32);
  }
  if (lane == 0) {
    float mean = s * (1.f / DIM);
    float var  = s2 * (1.f / DIM) - mean * mean;
    mu[row]   = mean;
    rstd[row] = rsqrtf(var + LN_EPS);
  }
}

// ---------------------------------------------------------------------------
// Kernel A: LN-apply + hi/lo bf16 split of H; hi/lo split of W (done once)
// ---------------------------------------------------------------------------
__global__ __launch_bounds__(256) void prep_kernel(
    const float* __restrict__ x, const float* __restrict__ gamma,
    const float* __restrict__ beta, const float* __restrict__ w,
    const float* __restrict__ mu, const float* __restrict__ rstd,
    unsigned short* __restrict__ Hhi, unsigned short* __restrict__ Hlo,
    unsigned short* __restrict__ Whi, unsigned short* __restrict__ Wlo) {
  int row = blockIdx.x;
  int t   = threadIdx.x;
  if (row < MROWS) {
    float mean = mu[row], rs = rstd[row];
    const float* xr = x + (size_t)row * DIM;
#pragma unroll
    for (int i = 0; i < 2; ++i) {
      int c = t + i * 256;
      float h = (xr[c] - mean) * rs * gamma[c] + beta[c];
      unsigned short hi, lo;
      bf_split(h, hi, lo);
      Hhi[(size_t)row * DIM + c] = hi;
      Hlo[(size_t)row * DIM + c] = lo;
    }
  } else {
    int r = row - MROWS;
    const float* wr = w + (size_t)r * DIM;
#pragma unroll
    for (int i = 0; i < 2; ++i) {
      int c = t + i * 256;
      unsigned short hi, lo;
      bf_split(wr[c], hi, lo);
      Whi[(size_t)r * DIM + c] = hi;
      Wlo[(size_t)r * DIM + c] = lo;
    }
  }
}

// ---------------------------------------------------------------------------
// Kernel 1: QKV projection, split-bf16:  C = Hhi*Whi + Hhi*Wlo + Hlo*Whi
// 64x64 tile / block (4 waves, wave = 16x64 strip).
// Q,K stored as hi/lo bf16 planes [bh][n][64]; V stored fp32.
// ---------------------------------------------------------------------------
__global__ __launch_bounds__(128) void qkv_gemm_kernel(
    const unsigned short* __restrict__ Hhi, const unsigned short* __restrict__ Hlo,
    const unsigned short* __restrict__ Whi, const unsigned short* __restrict__ Wlo,
    unsigned short* __restrict__ Qhi, unsigned short* __restrict__ Qlo,
    unsigned short* __restrict__ Khi, unsigned short* __restrict__ Klo,
    float* __restrict__ Vf) {
  __shared__ __align__(16) unsigned short Hs[2][64][LPITCH];
  __shared__ __align__(16) unsigned short Ws[2][64][LPITCH];

  const int m0   = blockIdx.x * 64;
  const int n0   = blockIdx.y * 64;
  const int tid  = threadIdx.x;
  const int wv   = tid >> 5;
  const int lane = tid & 31;
  const int half = lane >> 4;
  const int lp   = lane & 15;

  v8f acc[4] = {};

  const int srow = tid >> 1;
  const int scol = (tid & 1) * 32;

  for (int k0 = 0; k0 < DIM; k0 += 64) {
    __syncthreads();
#pragma unroll
    for (int g = 0; g < 4; ++g) {
      int c = scol + g * 8;
      *(v8us*)&Hs[0][srow][c] = *(const v8us*)(Hhi + (size_t)(m0 + srow) * DIM + k0 + c);
      *(v8us*)&Hs[1][srow][c] = *(const v8us*)(Hlo + (size_t)(m0 + srow) * DIM + k0 + c);
      *(v8us*)&Ws[0][srow][c] = *(const v8us*)(Whi + (size_t)(n0 + srow) * DIM + k0 + c);
      *(v8us*)&Ws[1][srow][c] = *(const v8us*)(Wlo + (size_t)(n0 + srow) * DIM + k0 + c);
    }
    __syncthreads();
#pragma unroll
    for (int kb = 0; kb < 64; kb += 32) {
      const unsigned short* ah = &Hs[0][wv * 16 + lp][kb];
      const unsigned short* al = &Hs[1][wv * 16 + lp][kb];
      v16us ahi = mk16(*(const v8us*)(ah + 8 * half), *(const v8us*)(ah + 16 + 8 * half));
      v16us alo = mk16(*(const v8us*)(al + 8 * half), *(const v8us*)(al + 16 + 8 * half));
#pragma unroll
      for (int s = 0; s < 4; ++s) {
        const unsigned short* bh = &Ws[0][s * 16 + lp][kb];
        const unsigned short* bl = &Ws[1][s * 16 + lp][kb];
        v16us bhi = mk16(*(const v8us*)(bh + 8 * half), *(const v8us*)(bh + 16 + 8 * half));
        v16us blo = mk16(*(const v8us*)(bl + 8 * half), *(const v8us*)(bl + 16 + 8 * half));
        acc[s] = wmma_bf16(ahi, bhi, acc[s]);
        acc[s] = wmma_bf16(ahi, blo, acc[s]);
        acc[s] = wmma_bf16(alo, bhi, acc[s]);
      }
    }
  }

  // Scatter: part/head uniform per block-column group (n0 is 64-aligned).
#pragma unroll
  for (int s = 0; s < 4; ++s) {
    int col  = n0 + s * 16 + lp;
    int part = col >> 9;
    int head = (col >> 6) & 7;
    int d    = col & 63;
#pragma unroll
    for (int r = 0; r < 8; ++r) {
      int row = m0 + wv * 16 + 8 * half + r;
      int b   = row >> 11;
      int ns  = row & (SEQ - 1);
      size_t off = (((size_t)(b * HEADS + head)) * SEQ + ns) * DHEAD + d;
      float v = acc[s][r];
      if (part == 0) {
        unsigned short hi, lo; bf_split(v, hi, lo);
        Qhi[off] = hi; Qlo[off] = lo;
      } else if (part == 1) {
        unsigned short hi, lo; bf_split(v, hi, lo);
        Khi[off] = hi; Klo[off] = lo;
      } else {
        Vf[off] = v;
      }
    }
  }
}

// ---------------------------------------------------------------------------
// Kernel 2: streaming attention, split-bf16 WMMA, reference 1e-8 mask fill
// (softmax over ALL 2048 columns). Block = (b*h, 64-row q tile), 4 waves.
// ---------------------------------------------------------------------------
__global__ __launch_bounds__(128) void attn_kernel(
    const unsigned short* __restrict__ Qhi, const unsigned short* __restrict__ Qlo,
    const unsigned short* __restrict__ Khi, const unsigned short* __restrict__ Klo,
    const float* __restrict__ Vf, float* __restrict__ out) {
  __shared__ __align__(16) unsigned short Ks[2][64][LPITCH];   // [plane][j][d]
  __shared__ __align__(16) unsigned short VTs[2][64][LPITCH];  // [plane][d][j] (transposed)
  __shared__ __align__(16) unsigned short Ps[2][4][16][LPITCH];// [plane][wave][i][j]

  const int bh   = blockIdx.x;
  const int qt   = blockIdx.y;
  const int b    = bh >> 3;
  const int head = bh & 7;
  const int tid  = threadIdx.x;
  const int wv   = tid >> 5;
  const int lane = tid & 31;
  const int half = lane >> 4;
  const int lp   = lane & 15;
  const int q0   = qt * 64;

  const size_t bho = (size_t)bh * SEQ * DHEAD;

  // Preload Q strip fragments (hi/lo), kept in VGPRs for the whole kernel.
  v16us aqh[2], aql[2];
  {
    const unsigned short* qh = Qhi + bho + (size_t)(q0 + wv * 16 + lp) * DHEAD;
    const unsigned short* ql = Qlo + bho + (size_t)(q0 + wv * 16 + lp) * DHEAD;
#pragma unroll
    for (int kb = 0; kb < 2; ++kb) {
      aqh[kb] = mk16(*(const v8us*)(qh + kb * 32 + 8 * half),
                     *(const v8us*)(qh + kb * 32 + 16 + 8 * half));
      aql[kb] = mk16(*(const v8us*)(ql + kb * 32 + 8 * half),
                     *(const v8us*)(ql + kb * 32 + 16 + 8 * half));
    }
  }

  v8f acc_o[4] = {};
  float mrow[8], lrow[8];
#pragma unroll
  for (int r = 0; r < 8; ++r) { mrow[r] = -1e30f; lrow[r] = 0.f; }

  const int srow = tid >> 1;
  const int scol = (tid & 1) * 32;

  for (int j0 = 0; j0 < SEQ; j0 += 64) {
    __syncthreads();
    {  // stage K (hi/lo, natural layout) and V (fp32 -> hi/lo, transposed)
      const unsigned short* kh = Khi + bho + (size_t)(j0 + srow) * DHEAD;
      const unsigned short* kl = Klo + bho + (size_t)(j0 + srow) * DHEAD;
#pragma unroll
      for (int g = 0; g < 4; ++g) {
        int c = scol + g * 8;
        *(v8us*)&Ks[0][srow][c] = *(const v8us*)(kh + c);
        *(v8us*)&Ks[1][srow][c] = *(const v8us*)(kl + c);
      }
      const float* vr = Vf + bho + (size_t)(j0 + srow) * DHEAD;
#pragma unroll
      for (int g = 0; g < 8; ++g) {
        float4 v4 = *(const float4*)(vr + scol + g * 4);
        float fv[4] = {v4.x, v4.y, v4.z, v4.w};
#pragma unroll
        for (int e = 0; e < 4; ++e) {
          unsigned short hi, lo;
          bf_split(fv[e], hi, lo);
          int d = scol + g * 4 + e;
          VTs[0][d][srow] = hi;
          VTs[1][d][srow] = lo;
        }
      }
    }
    __syncthreads();

    // ---- S = Q K^T (strip 16 x 64): 2 K-blocks x 4 subtiles x 3 WMMA ----
    v8f s[4] = {};
#pragma unroll
    for (int kb = 0; kb < 2; ++kb) {
#pragma unroll
      for (int nj = 0; nj < 4; ++nj) {
        const unsigned short* bh16 = &Ks[0][nj * 16 + lp][kb * 32];
        const unsigned short* bl16 = &Ks[1][nj * 16 + lp][kb * 32];
        v16us bhi = mk16(*(const v8us*)(bh16 + 8 * half), *(const v8us*)(bh16 + 16 + 8 * half));
        v16us blo = mk16(*(const v8us*)(bl16 + 8 * half), *(const v8us*)(bl16 + 16 + 8 * half));
        s[nj] = wmma_bf16(aqh[kb], bhi, s[nj]);
        s[nj] = wmma_bf16(aqh[kb], blo, s[nj]);
        s[nj] = wmma_bf16(aql[kb], bhi, s[nj]);
      }
    }

    // ---- mask (1e-8 fill) + online softmax; write P hi/lo to LDS ----
#pragma unroll
    for (int r = 0; r < 8; ++r) {
      int i = q0 + wv * 16 + 8 * half + r;
      float sv[4];
#pragma unroll
      for (int nj = 0; nj < 4; ++nj) {
        int j = j0 + nj * 16 + lp;
        sv[nj] = (j <= i) ? s[nj][r] : MASK_FILL;
      }
      float cmax = fmaxf(fmaxf(sv[0], sv[1]), fmaxf(sv[2], sv[3]));
#pragma unroll
      for (int m = 8; m >= 1; m >>= 1) cmax = fmaxf(cmax, __shfl_xor(cmax, m, 16));
      float mnew  = fmaxf(mrow[r], cmax);
      float alpha = __expf(mrow[r] - mnew);
      float psum  = 0.f;
#pragma unroll
      for (int nj = 0; nj < 4; ++nj) {
        float p = __expf(sv[nj] - mnew);
        unsigned short phi, plo;
        bf_split(p, phi, plo);
        Ps[0][wv][r + 8 * half][nj * 16 + lp] = phi;
        Ps[1][wv][r + 8 * half][nj * 16 + lp] = plo;
        psum += p;
      }
#pragma unroll
      for (int m = 8; m >= 1; m >>= 1) psum += __shfl_xor(psum, m, 16);
      lrow[r] = lrow[r] * alpha + psum;
      mrow[r] = mnew;
#pragma unroll
      for (int nd = 0; nd < 4; ++nd) acc_o[nd][r] *= alpha;
    }

    // Per-wave LDS round-trip (D-frag -> A-frag relayout); LDS in-order/wave.
    asm volatile("s_wait_dscnt 0" ::: "memory");

    // ---- O += P * V: 2 K-blocks x 4 subtiles x 3 WMMA ----
#pragma unroll
    for (int kb = 0; kb < 2; ++kb) {
      const unsigned short* ph = &Ps[0][wv][lp][kb * 32];
      const unsigned short* pl = &Ps[1][wv][lp][kb * 32];
      v16us pah = mk16(*(const v8us*)(ph + 8 * half), *(const v8us*)(ph + 16 + 8 * half));
      v16us pal = mk16(*(const v8us*)(pl + 8 * half), *(const v8us*)(pl + 16 + 8 * half));
#pragma unroll
      for (int nd = 0; nd < 4; ++nd) {
        const unsigned short* vh = &VTs[0][nd * 16 + lp][kb * 32];
        const unsigned short* vl = &VTs[1][nd * 16 + lp][kb * 32];
        v16us bvh = mk16(*(const v8us*)(vh + 8 * half), *(const v8us*)(vh + 16 + 8 * half));
        v16us bvl = mk16(*(const v8us*)(vl + 8 * half), *(const v8us*)(vl + 16 + 8 * half));
        acc_o[nd] = wmma_bf16(pah, bvh, acc_o[nd]);
        acc_o[nd] = wmma_bf16(pah, bvl, acc_o[nd]);
        acc_o[nd] = wmma_bf16(pal, bvh, acc_o[nd]);
      }
    }
  }

  // ---- normalize and store: out[b][i][head*64 + d] ----
#pragma unroll
  for (int r = 0; r < 8; ++r) {
    int i = q0 + wv * 16 + 8 * half + r;
    float inv = 1.f / lrow[r];
#pragma unroll
    for (int nd = 0; nd < 4; ++nd) {
      size_t off = ((size_t)b * SEQ + i) * DIM + head * DHEAD + nd * 16 + lp;
      out[off] = acc_o[nd][r] * inv;
    }
  }
}

// ---------------------------------------------------------------------------
// Launch
// ---------------------------------------------------------------------------
extern "C" void kernel_launch(void* const* d_in, const int* in_sizes, int n_in,
                              void* d_out, int out_size, void* d_ws, size_t ws_size,
                              hipStream_t stream) {
  const float* x     = (const float*)d_in[0];
  const float* gamma = (const float*)d_in[1];
  const float* beta  = (const float*)d_in[2];
  const float* w_qkv = (const float*)d_in[3];
  // d_in[4] tril mask: causality applied analytically in-kernel.
  float* out = (float*)d_out;

  char* p = (char*)d_ws;
  float* mu   = (float*)p; p += (size_t)MROWS * 4;
  float* rstd = (float*)p; p += (size_t)MROWS * 4;
  const size_t hplane = (size_t)MROWS * DIM;       // bf16 elems
  const size_t wplane = (size_t)QKV3 * DIM;
  const size_t qplane = (size_t)BATCH * HEADS * SEQ * DHEAD;
  unsigned short* Hhi = (unsigned short*)p; p += hplane * 2;
  unsigned short* Hlo = (unsigned short*)p; p += hplane * 2;
  unsigned short* Whi = (unsigned short*)p; p += wplane * 2;
  unsigned short* Wlo = (unsigned short*)p; p += wplane * 2;
  unsigned short* Qhi = (unsigned short*)p; p += qplane * 2;
  unsigned short* Qlo = (unsigned short*)p; p += qplane * 2;
  unsigned short* Khi = (unsigned short*)p; p += qplane * 2;
  unsigned short* Klo = (unsigned short*)p; p += qplane * 2;
  float* Vf = (float*)p;

  ln_stats_kernel<<<MROWS / 8, 256, 0, stream>>>(x, mu, rstd);

  prep_kernel<<<MROWS + QKV3, 256, 0, stream>>>(x, gamma, beta, w_qkv, mu, rstd,
                                                Hhi, Hlo, Whi, Wlo);

  dim3 g1(MROWS / 64, QKV3 / 64);   // 128 x 24
  qkv_gemm_kernel<<<g1, 128, 0, stream>>>(Hhi, Hlo, Whi, Wlo,
                                          Qhi, Qlo, Khi, Klo, Vf);

  dim3 g2(BATCH * HEADS, SEQ / 64); // 32 x 32
  attn_kernel<<<g2, 128, 0, stream>>>(Qhi, Qlo, Khi, Klo, Vf, out);
}
